// SGDATSeg_61409442398669
// MI455X (gfx1250) — matrix operand order repaired and checked by reading, not compiled, not run on hardware
//
#include <hip/hip_runtime.h>

// ---------------- problem constants ----------------
#define BB   8
#define NPTS 8000
#define CH   64
#define MM1  512
#define MM2  128

typedef __attribute__((ext_vector_type(16))) _Float16 v16h;
typedef __attribute__((ext_vector_type(8)))  float    v8f;

// ---------------- WMMA helpers (CDNA5 16x16x32 f16, wave32) ----------------
__device__ __forceinline__ v8f v8f_zero() {
  v8f z;
#pragma unroll
  for (int j = 0; j < 8; ++j) z[j] = 0.f;
  return z;
}

__device__ __forceinline__ v8f wmma16(v16h a, v16h b, v8f c) {
  return __builtin_amdgcn_wmma_f32_16x16x32_f16(false, a, false, b, (short)0, c,
                                                false, false);
}

// A fragment: 16x32 f16 from row-major LDS A[16][lda]; per-lane elements are one
// contiguous 16-half run (k = 16h + e).
__device__ __forceinline__ v16h frag_a(const _Float16* __restrict__ A, int lda, int lane) {
  v16h r;
  const int m  = lane & 15;
  const int kb = lane & 16;                      // 0 or 16
  const _Float16* row = A + m * lda;
#pragma unroll
  for (int e = 0; e < 16; ++e) {
    const int k = kb + e;                        // contiguous k order, matches B
    r[e] = row[k];
  }
  return r;
}

// B fragment from pre-swizzled LDS: one aligned 32-byte load per lane.
__device__ __forceinline__ v16h frag_b_sw(const _Float16* __restrict__ Wsw,
                                          int blk, int lane) {
  return *(const v16h*)(Wsw + (((blk << 5) + lane) << 4));
}

// 16 x (NT*16) = A(16 x KC*32) @ W(swizzled), A f16 row-major in LDS
template <int KC, int NT>
__device__ __forceinline__ void tile_gemm_sw(const _Float16* __restrict__ A, int lda,
                                             const _Float16* __restrict__ Wsw,
                                             v8f* acc, int lane) {
#pragma unroll
  for (int t = 0; t < NT; ++t) acc[t] = v8f_zero();
#pragma unroll
  for (int kc = 0; kc < KC; ++kc) {
    v16h a = frag_a(A + kc * 32, lda, lane);
#pragma unroll
    for (int t = 0; t < NT; ++t) {
      v16h b = frag_b_sw(Wsw, kc * NT + t, lane);
      acc[t] = wmma16(a, b, acc[t]);
    }
  }
}

// One-time scatter of f32 weight (Korig x Ncols row-major) into B-fragment order,
// zero-padded to (KC*32) x (NT*16).
__device__ __forceinline__ void load_w_sw(_Float16* dst, const float* src,
                                          int Korig, int Ncols, int KC, int NT,
                                          int tid, int nthr) {
  const int total = KC * NT * 512;
  for (int i = tid; i < total; i += nthr) {
    const int e = i & 15, ln = (i >> 4) & 31, blk = i >> 9;
    const int t = blk % NT, kc = blk / NT;
    const int kr = kc * 32 + (ln & 16) + e;
    const int nc = t * 16 + (ln & 15);
    dst[i] = (_Float16)((kr < Korig && nc < Ncols) ? src[kr * Ncols + nc] : 0.f);
  }
}

// Packed f32 -> f16 tile staging: float4 global loads, 8-byte LDS stores.
__device__ __forceinline__ void stage_f16(_Float16* dst, const float* src,
                                          int nelem, int tid, int nthr) {
  const float4* s4 = (const float4*)src;
  uint2* d2 = (uint2*)dst;
  for (int i = tid; i < (nelem >> 2); i += nthr) {
    const float4 v = s4[i];
    union { _Float16 h[4]; uint2 u; } pk;
    pk.h[0] = (_Float16)v.x; pk.h[1] = (_Float16)v.y;
    pk.h[2] = (_Float16)v.z; pk.h[3] = (_Float16)v.w;
    d2[i] = pk.u;
  }
}

// ---------------- kernel 1: encoder  feats = relu(x@W1+b1)@W2+b2 --------------
__global__ void __launch_bounds__(128)
k_encode(const float* __restrict__ x, const float* __restrict__ w1,
         const float* __restrict__ b1, const float* __restrict__ w2,
         const float* __restrict__ b2, float* __restrict__ feats) {
  __shared__ __align__(32) _Float16 sW2[2 * 4 * 512];
  __shared__ float sW1[9 * 64];
  __shared__ float sB1[64], sB2[64];
  __shared__ float sX[4][16 * 9];
  __shared__ __align__(32) _Float16 sA[4][16 * 64];

  const int tid = threadIdx.x;
  load_w_sw(sW2, w2, 64, 64, 2, 4, tid, 128);
  for (int i = tid; i < 9 * 64; i += 128) sW1[i] = w1[i];
  for (int i = tid; i < 64; i += 128) { sB1[i] = b1[i]; sB2[i] = b2[i]; }

  const int wave = tid >> 5, lane = tid & 31;
  const int r0 = (blockIdx.x * 4 + wave) * 16;   // flat row over B*N
  for (int i = lane; i < 144; i += 32) sX[wave][i] = x[(size_t)r0 * 9 + i];
  __syncthreads();

  // layer 1 (K=9): each lane computes 2 adjacent channels per row, packed store
  const int c0 = lane * 2;
  for (int rr = 0; rr < 16; ++rr) {
    float a0 = sB1[c0], a1 = sB1[c0 + 1];
#pragma unroll
    for (int k = 0; k < 9; ++k) {
      const float xv = sX[wave][rr * 9 + k];
      a0 += xv * sW1[k * 64 + c0];
      a1 += xv * sW1[k * 64 + c0 + 1];
    }
    union { _Float16 h[2]; unsigned u; } pk;
    pk.h[0] = (_Float16)fmaxf(a0, 0.f);
    pk.h[1] = (_Float16)fmaxf(a1, 0.f);
    ((unsigned*)sA[wave])[rr * 32 + lane] = pk.u;
  }

  v8f acc[4];
  tile_gemm_sw<2, 4>(sA[wave], 64, sW2, acc, lane);
  const int n = lane & 15, mh = (lane & 16) >> 1;
#pragma unroll
  for (int t = 0; t < 4; ++t)
#pragma unroll
    for (int j = 0; j < 8; ++j) {
      const int ch = t * 16 + n;
      feats[(size_t)(r0 + j + mh) * 64 + ch] = acc[t][j] + sB2[ch];
    }
}

// ---------------- kernel 2: farthest point sampling (512 steps) ---------------
__global__ void __launch_bounds__(1024)
k_fps(const float* __restrict__ x, int* __restrict__ idx) {
  const int b = blockIdx.x, tid = threadIdx.x;
  float cx[8], cy[8], cz[8], md[8];
#pragma unroll
  for (int s = 0; s < 8; ++s) {
    const int p = tid + s * 1024;
    if (p < NPTS) {
      const float* xr = x + ((size_t)b * NPTS + p) * 9;
      cx[s] = xr[0]; cy[s] = xr[1]; cz[s] = xr[2]; md[s] = 1e10f;
    } else { cx[s] = 0.f; cy[s] = 0.f; cz[s] = 0.f; md[s] = -1.f; }
  }
  __shared__ float sbc[3];
  __shared__ float swv[32];
  __shared__ int   swi[32];
  __shared__ int   slast;
  int last = 0;
  for (int it = 0; it < MM1; ++it) {
    if (tid == 0) idx[b * MM1 + it] = last;
    if ((last & 1023) == tid) {
      const int s = last >> 10;
      sbc[0] = cx[s]; sbc[1] = cy[s]; sbc[2] = cz[s];
    }
    __syncthreads();
    const float lx = sbc[0], ly = sbc[1], lz = sbc[2];
    float bv = -2.f; int bi = 0;
#pragma unroll
    for (int s = 0; s < 8; ++s) {
      const int p = tid + s * 1024;
      const float dx = cx[s] - lx, dy = cy[s] - ly, dz = cz[s] - lz;
      float m = fminf(md[s], dx * dx + dy * dy + dz * dz);
      if (p < NPTS) md[s] = m; else m = -1.f;
      if (m > bv) { bv = m; bi = p; }
    }
#pragma unroll
    for (int off = 16; off > 0; off >>= 1) {
      const float ov = __shfl_xor(bv, off, 32);
      const int   oi = __shfl_xor(bi, off, 32);
      if (ov > bv || (ov == bv && oi < bi)) { bv = ov; bi = oi; }
    }
    if ((tid & 31) == 0) { swv[tid >> 5] = bv; swi[tid >> 5] = bi; }
    __syncthreads();
    if (tid < 32) {
      bv = swv[tid]; bi = swi[tid];
#pragma unroll
      for (int off = 16; off > 0; off >>= 1) {
        const float ov = __shfl_xor(bv, off, 32);
        const int   oi = __shfl_xor(bi, off, 32);
        if (ov > bv || (ov == bv && oi < bi)) { bv = ov; bi = oi; }
      }
      if (tid == 0) slast = bi;
    }
    __syncthreads();
    last = slast;
  }
}

// ---------------- kernel 3: kNN grouping + masked-max MLP (per center) --------
__global__ void __launch_bounds__(64)
k_fuse(const float* __restrict__ x, const float* __restrict__ feats,
       const int* __restrict__ idx, const float* __restrict__ W,
       const float* __restrict__ bias, float* __restrict__ out, int M,
       float rmin, float rmax) {
  const int bm = blockIdx.x, b = bm / M, m = bm % M, tid = threadIdx.x;
  __shared__ __align__(32) _Float16 sW[3 * 4 * 512];   // f1_w padded 67->96 rows
  __shared__ float sb[64];
  __shared__ float cd[64 * 16];
  __shared__ int   ci[64 * 16];
  __shared__ float sknn[16];
  __shared__ int   sknni[16];
  __shared__ int   smask[16];
  __shared__ __align__(32) _Float16 sA[16 * 96];
  __shared__ float sc[3];
  __shared__ float rv[64];
  __shared__ int   ri[64], rs[64];

  load_w_sw(sW, W, 67, 64, 3, 4, tid, 64);
  if (tid < 64) sb[tid] = bias[tid];
  if (tid == 0) {
    const int ctr = idx[b * MM1 + m];            // idx2 == idx1[:, :128]
    const float* xr = x + ((size_t)b * NPTS + ctr) * 9;
    sc[0] = xr[0]; sc[1] = xr[1]; sc[2] = xr[2];
  }
  __syncthreads();
  const float ccx = sc[0], ccy = sc[1], ccz = sc[2];

  // per-thread top-16 (smallest d2, tie -> lower index)
  float ld[16]; int li[16];
#pragma unroll
  for (int j = 0; j < 16; ++j) { ld[j] = 1e30f; li[j] = 0x7fffffff; }
  for (int p = tid; p < NPTS; p += 64) {
    const float* xr = x + ((size_t)b * NPTS + p) * 9;
    const float dx = xr[0] - ccx, dy = xr[1] - ccy, dz = xr[2] - ccz;
    const float d = dx * dx + dy * dy + dz * dz;
    if (d < ld[15] || (d == ld[15] && p < li[15])) {
      int j = 15;
      while (j > 0 && (d < ld[j - 1] || (d == ld[j - 1] && p < li[j - 1]))) {
        ld[j] = ld[j - 1]; li[j] = li[j - 1]; --j;
      }
      ld[j] = d; li[j] = p;
    }
  }
#pragma unroll
  for (int j = 0; j < 16; ++j) { cd[tid * 16 + j] = ld[j]; ci[tid * 16 + j] = li[j]; }
  __syncthreads();

  // merge: 16 rounds of global argmin over 1024 candidates
  for (int r = 0; r < 16; ++r) {
    float bv = 1e30f; int bi = 0x7fffffff, bs = -1;
    for (int s2 = tid; s2 < 1024; s2 += 64) {
      const float vv = cd[s2]; const int ii = ci[s2];
      if (vv < bv || (vv == bv && ii < bi)) { bv = vv; bi = ii; bs = s2; }
    }
    rv[tid] = bv; ri[tid] = bi; rs[tid] = bs;
    __syncthreads();
    for (int off = 32; off > 0; off >>= 1) {
      if (tid < off) {
        const float ov = rv[tid + off]; const int oi = ri[tid + off];
        if (ov < rv[tid] || (ov == rv[tid] && oi < ri[tid])) {
          rv[tid] = ov; ri[tid] = oi; rs[tid] = rs[tid + off];
        }
      }
      __syncthreads();
    }
    if (tid == 0) {
      sknn[r] = sqrtf(fmaxf(rv[0], 0.f));
      sknni[r] = ri[0];
      cd[rs[0]] = 1e30f;
    }
    __syncthreads();
  }
  if (tid == 0) {
    float s = 0.f;
#pragma unroll
    for (int j = 0; j < 16; ++j) s += sknn[j];
    const float rad = fminf(fmaxf(s * (1.f / 16.f), rmin), rmax);
#pragma unroll
    for (int j = 0; j < 16; ++j) smask[j] = (sknn[j] <= rad) ? 1 : 0;
  }
  __syncthreads();

  // build A = [feats | rel_xyz | 0pad] (16 x 96) in f16
  for (int i = tid; i < 16 * 96; i += 64) {
    const int rrow = i / 96, cc = i % 96;
    const int pidx = sknni[rrow];
    float v;
    if (cc < 64) v = feats[((size_t)b * NPTS + pidx) * 64 + cc];
    else if (cc < 67) {
      const float* xr = x + ((size_t)b * NPTS + pidx) * 9;
      v = xr[cc - 64] - sc[cc - 64];
    } else v = 0.f;
    sA[rrow * 96 + cc] = (_Float16)v;
  }
  __syncthreads();

  if (tid < 32) {                                 // wave 0: WMMA + masked max
    const int lane = tid;
    v8f acc[4];
    tile_gemm_sw<3, 4>(sA, 96, sW, acc, lane);
    const int n = lane & 15, mh = (lane & 16) >> 1;
#pragma unroll
    for (int t = 0; t < 4; ++t) {
      const int ch = t * 16 + n;
      const float bb = sb[ch];
      float mx = -1e9f;
#pragma unroll
      for (int j = 0; j < 8; ++j) {
        const float h = fmaxf(acc[t][j] + bb, 0.f);
        mx = fmaxf(mx, smask[j + mh] ? h : -1e9f);
      }
      mx = fmaxf(mx, __shfl_xor(mx, 16, 32));
      if (lane < 16) out[((size_t)(b * M + m)) * 64 + ch] = mx;
    }
  }
}

// ---------------- kernel 4: ChannelCCC (squeeze-excite), in place -------------
__global__ void __launch_bounds__(64)
k_ccc(float* __restrict__ xio, const float* __restrict__ w1,
      const float* __restrict__ b1, const float* __restrict__ w2,
      const float* __restrict__ b2, int M) {
  const int b = blockIdx.x, tid = threadIdx.x;
  __shared__ float mean[64], hid[16], sS[64];
  float* xb = xio + (size_t)b * M * 64;
  float acc = 0.f;
  for (int r = 0; r < M; ++r) acc += xb[r * 64 + tid];
  mean[tid] = acc / (float)M;
  __syncthreads();
  if (tid < 16) {
    float a = b1[tid];
    for (int c = 0; c < 64; ++c) a += mean[c] * w1[c * 16 + tid];
    hid[tid] = fmaxf(a, 0.f);
  }
  __syncthreads();
  {
    float a = b2[tid];
#pragma unroll
    for (int h = 0; h < 16; ++h) a += hid[h] * w2[h * 64 + tid];
    sS[tid] = 1.f / (1.f + __expf(-a));
  }
  __syncthreads();
  for (int i = tid; i < M * 64; i += 64) xb[i] *= sS[i & 63];
}

// ---------------- kernel 5: q/k/v gemms (elu+1 on q,k) ------------------------
__global__ void __launch_bounds__(128)
k_qkv(const float* __restrict__ xin, const float* __restrict__ wq,
      const float* __restrict__ wk, const float* __restrict__ wv,
      float* __restrict__ q, float* __restrict__ k, float* __restrict__ v) {
  __shared__ __align__(32) _Float16 sWq[4096], sWk[4096], sWv[4096];
  __shared__ __align__(32) _Float16 sA[4][16 * 64];
  const int tid = threadIdx.x;
  load_w_sw(sWq, wq, 64, 64, 2, 4, tid, 128);
  load_w_sw(sWk, wk, 64, 64, 2, 4, tid, 128);
  load_w_sw(sWv, wv, 64, 64, 2, 4, tid, 128);
  const int wave = tid >> 5, lane = tid & 31;
  const int r0 = (blockIdx.x * 4 + wave) * 16;
  stage_f16(sA[wave], xin + (size_t)r0 * 64, 1024, lane, 32);
  __syncthreads();
  v8f aq[4], ak[4], av[4];
  tile_gemm_sw<2, 4>(sA[wave], 64, sWq, aq, lane);
  tile_gemm_sw<2, 4>(sA[wave], 64, sWk, ak, lane);
  tile_gemm_sw<2, 4>(sA[wave], 64, sWv, av, lane);
  const int n = lane & 15, mh = (lane & 16) >> 1;
#pragma unroll
  for (int t = 0; t < 4; ++t)
#pragma unroll
    for (int j = 0; j < 8; ++j) {
      const size_t o = (size_t)(r0 + j + mh) * 64 + t * 16 + n;
      float qq = aq[t][j]; qq = (qq > 0.f) ? qq + 1.f : __expf(qq);
      float kk = ak[t][j]; kk = (kk > 0.f) ? kk + 1.f : __expf(kk);
      q[o] = qq; k[o] = kk; v[o] = av[t][j];
    }
}

// ---------------- kernel 6: kv = k^T v (per batch) + ksum ---------------------
__global__ void __launch_bounds__(128)
k_kvsum(const float* __restrict__ kg, const float* __restrict__ vg,
        float* __restrict__ kv, float* __restrict__ ksum, int M) {
  const int b = blockIdx.x, tid = threadIdx.x;
  const int wave = tid >> 5, lane = tid & 31;
  const float* kb = kg + (size_t)b * M * 64;
  const float* vb = vg + (size_t)b * M * 64;
  __shared__ __align__(32) _Float16 sKT[64 * 32];      // k^T tile (c-major)
  __shared__ __align__(32) _Float16 sVsw[4 * 512];     // v chunk, B-swizzled
  __shared__ float sks[128];

  // ksum in f32 (two half-sums per channel, fixed combine order)
  {
    const int c = tid & 63, half = tid >> 6;
    const int mh2 = M >> 1;
    float s = 0.f;
    for (int r = half * mh2; r < (half + 1) * mh2; ++r) s += kb[(size_t)r * 64 + c];
    sks[tid] = s;
  }

  const int ct = wave;                                 // c-tile per wave (4 waves)
  v8f acc[4];
#pragma unroll
  for (int t = 0; t < 4; ++t) acc[t] = v8f_zero();

  for (int n0 = 0; n0 < M; n0 += 32) {
    __syncthreads();
    for (int i = tid; i < 2048; i += 128) {            // stage k^T (coalesced reads)
      const int n = i >> 6, c = i & 63;
      sKT[c * 32 + n] = (_Float16)kb[(size_t)(n0 + n) * 64 + c];
    }
    for (int i = tid; i < 2048; i += 128) {            // stage v, swizzled
      const int e = i & 15, ln = (i >> 4) & 31, t = i >> 9;
      sVsw[i] = (_Float16)vb[(size_t)(n0 + (ln & 16) + e) * 64 + t * 16 + (ln & 15)];
    }
    __syncthreads();
    v16h a = frag_a(sKT + ct * 16 * 32, 32, lane);
#pragma unroll
    for (int t = 0; t < 4; ++t)
      acc[t] = wmma16(a, frag_b_sw(sVsw, t, lane), acc[t]);
  }
  const int n = lane & 15, mh = (lane & 16) >> 1;
#pragma unroll
  for (int t = 0; t < 4; ++t)
#pragma unroll
    for (int j = 0; j < 8; ++j)
      kv[((size_t)b * 64 + ct * 16 + j + mh) * 64 + t * 16 + n] = acc[t][j];
  if (tid < 64) ksum[b * 64 + tid] = sks[tid] + sks[tid + 64];
}

// ---------------- kernel 7: out = x + ((q@kv)/z)@wo ---------------------------
__global__ void __launch_bounds__(128)
k_attnout(const float* __restrict__ xin, const float* __restrict__ q,
          const float* __restrict__ kv, const float* __restrict__ ksum,
          const float* __restrict__ wo, float* __restrict__ gout, int M) {
  const int b = blockIdx.x, tid = threadIdx.x;
  const int wave = tid >> 5, lane = tid & 31;
  __shared__ __align__(32) _Float16 sKV[4096], sWo[4096];
  __shared__ __align__(32) _Float16 sQ[4][16 * 64], sU[4][16 * 64];
  __shared__ float sZ[4][16];
  load_w_sw(sKV, kv + (size_t)b * 4096, 64, 64, 2, 4, tid, 128);
  load_w_sw(sWo, wo, 64, 64, 2, 4, tid, 128);
  const int m0 = (blockIdx.y * 4 + wave) * 16;
  const float* qb = q + (size_t)b * M * 64;
  stage_f16(sQ[wave], qb + (size_t)m0 * 64, 1024, lane, 32);
  if (lane < 16) {
    const float* qr = qb + (size_t)(m0 + lane) * 64;
    const float* ks = ksum + b * 64;
    float z = 0.f;
#pragma unroll
    for (int c = 0; c < 64; ++c) z += qr[c] * ks[c];
    sZ[wave][lane] = 1.f / (z + 1e-6f);
  }
  __syncthreads();
  v8f u[4];
  tile_gemm_sw<2, 4>(sQ[wave], 64, sKV, u, lane);
  const int n = lane & 15, mh = (lane & 16) >> 1;
#pragma unroll
  for (int t = 0; t < 4; ++t)
#pragma unroll
    for (int j = 0; j < 8; ++j)
      sU[wave][(j + mh) * 64 + t * 16 + n] = (_Float16)(u[t][j] * sZ[wave][j + mh]);
  v8f o[4];
  tile_gemm_sw<2, 4>(sU[wave], 64, sWo, o, lane);
#pragma unroll
  for (int t = 0; t < 4; ++t)
#pragma unroll
    for (int j = 0; j < 8; ++j) {
      const size_t off = ((size_t)b * M + m0 + j + mh) * 64 + t * 16 + n;
      gout[off] = xin[off] + o[t][j];
    }
}

// ---------------- kernel 8: upsample-concat + head (128->64->64->32->13) ------
__global__ void __launch_bounds__(128)
k_final(const float* __restrict__ g1, const float* __restrict__ g2,
        const float* __restrict__ upw1, const float* __restrict__ upb1,
        const float* __restrict__ upw2, const float* __restrict__ upb2,
        const float* __restrict__ cw1, const float* __restrict__ cb1,
        const float* __restrict__ cw2, const float* __restrict__ cb2,
        float* __restrict__ out) {
  __shared__ __align__(32) _Float16 sU1[4 * 4 * 512], sU2[2 * 4 * 512],
                                    sC1[2 * 2 * 512], sC2[512];
  __shared__ float sb1[64], sb2[64], sb3[32], sb4[16];
  __shared__ __align__(32) _Float16 sA[4][16 * 128];
  const int tid = threadIdx.x;
  load_w_sw(sU1, upw1, 128, 64, 4, 4, tid, 128);
  load_w_sw(sU2, upw2, 64, 64, 2, 4, tid, 128);
  load_w_sw(sC1, cw1, 64, 32, 2, 2, tid, 128);
  load_w_sw(sC2, cw2, 32, 13, 1, 1, tid, 128);
  if (tid < 64) { sb1[tid] = upb1[tid]; sb2[tid] = upb2[tid]; }
  if (tid < 32) sb3[tid] = cb1[tid];
  if (tid < 16) sb4[tid] = (tid < 13) ? cb2[tid] : 0.f;

  const int wave = tid >> 5, lane = tid & 31;
  const int r0 = (blockIdx.x * 4 + wave) * 16;   // flat row over B*N
  // gather + pack input tile [16 x 128] (g1 | g2 upsampled)
  for (int qd = lane; qd < 512; qd += 32) {
    const int rr = qd >> 5, cq = (qd & 31) << 2;
    const int row = r0 + rr, bb2 = row / NPTS, p = row % NPTS;
    const float* srcp;
    if (cq < 64) srcp = g1 + ((size_t)bb2 * MM1 + (p * MM1) / NPTS) * 64 + cq;
    else         srcp = g2 + ((size_t)bb2 * MM2 + (p * MM2) / NPTS) * 64 + (cq - 64);
    const float4 v = *(const float4*)srcp;
    union { _Float16 h[4]; uint2 u; } pk;
    pk.h[0] = (_Float16)v.x; pk.h[1] = (_Float16)v.y;
    pk.h[2] = (_Float16)v.z; pk.h[3] = (_Float16)v.w;
    ((uint2*)sA[wave])[qd] = pk.u;
  }
  __syncthreads();
  const int n = lane & 15, mh = (lane & 16) >> 1;

  v8f a1[4];
  tile_gemm_sw<4, 4>(sA[wave], 128, sU1, a1, lane);    // 128 -> 64, relu
#pragma unroll
  for (int t = 0; t < 4; ++t)
#pragma unroll
    for (int j = 0; j < 8; ++j)
      sA[wave][(j + mh) * 64 + t * 16 + n] =
          (_Float16)fmaxf(a1[t][j] + sb1[t * 16 + n], 0.f);

  v8f a2[4];
  tile_gemm_sw<2, 4>(sA[wave], 64, sU2, a2, lane);     // 64 -> 64, no relu
#pragma unroll
  for (int t = 0; t < 4; ++t)
#pragma unroll
    for (int j = 0; j < 8; ++j)
      sA[wave][(j + mh) * 64 + t * 16 + n] =
          (_Float16)(a2[t][j] + sb2[t * 16 + n]);

  v8f a3[2];
  tile_gemm_sw<2, 2>(sA[wave], 64, sC1, a3, lane);     // 64 -> 32, relu
#pragma unroll
  for (int t = 0; t < 2; ++t)
#pragma unroll
    for (int j = 0; j < 8; ++j)
      sA[wave][(j + mh) * 32 + t * 16 + n] =
          (_Float16)fmaxf(a3[t][j] + sb3[t * 16 + n], 0.f);

  v8f a4[1];
  tile_gemm_sw<1, 1>(sA[wave], 32, sC2, a4, lane);     // 32 -> 16 (13 valid)
  if (n < 13) {
#pragma unroll
    for (int j = 0; j < 8; ++j)
      out[(size_t)(r0 + j + mh) * 13 + n] = a4[0][j] + sb4[n];
  }
}

// ---------------- host launch --------------------------------------------------
extern "C" void kernel_launch(void* const* d_in, const int* in_sizes, int n_in,
                              void* d_out, int out_size, void* d_ws, size_t ws_size,
                              hipStream_t stream) {
  (void)in_sizes; (void)n_in; (void)out_size; (void)ws_size;
  const float* x      = (const float*)d_in[0];
  const float* enc_w1 = (const float*)d_in[1];
  const float* enc_b1 = (const float*)d_in[2];
  const float* enc_w2 = (const float*)d_in[3];
  const float* enc_b2 = (const float*)d_in[4];
  const float* f1_w   = (const float*)d_in[5];
  const float* f1_b   = (const float*)d_in[6];
  const float* f2_w   = (const float*)d_in[7];
  const float* f2_b   = (const float*)d_in[8];
  const float* ccc_w1 = (const float*)d_in[9];
  const float* ccc_b1 = (const float*)d_in[10];
  const float* ccc_w2 = (const float*)d_in[11];
  const float* ccc_b2 = (const float*)d_in[12];
  const float* wq     = (const float*)d_in[13];
  const float* wk     = (const float*)d_in[14];
  const float* wv     = (const float*)d_in[15];
  const float* wo     = (const float*)d_in[16];
  const float* up_w1  = (const float*)d_in[17];
  const float* up_b1  = (const float*)d_in[18];
  const float* up_w2  = (const float*)d_in[19];
  const float* up_b2  = (const float*)d_in[20];
  const float* cls_w1 = (const float*)d_in[21];
  const float* cls_b1 = (const float*)d_in[22];
  const float* cls_w2 = (const float*)d_in[23];
  const float* cls_b2 = (const float*)d_in[24];
  float* out = (float*)d_out;

  char* base = (char*)d_ws;
  size_t off = 0;
  auto alloc = [&](size_t bytes) -> void* {
    void* p = base + off;
    off = (off + bytes + 255) & ~(size_t)255;
    return p;
  };
  float* feats = (float*)alloc((size_t)BB * NPTS * CH * 4);
  int*   idx1  = (int*)  alloc((size_t)BB * MM1 * 4);
  float* out1  = (float*)alloc((size_t)BB * MM1 * CH * 4);
  float* out2  = (float*)alloc((size_t)BB * MM2 * CH * 4);
  float* q1    = (float*)alloc((size_t)BB * MM1 * CH * 4);
  float* k1    = (float*)alloc((size_t)BB * MM1 * CH * 4);
  float* v1    = (float*)alloc((size_t)BB * MM1 * CH * 4);
  float* q2    = (float*)alloc((size_t)BB * MM2 * CH * 4);
  float* k2    = (float*)alloc((size_t)BB * MM2 * CH * 4);
  float* v2    = (float*)alloc((size_t)BB * MM2 * CH * 4);
  float* kv1   = (float*)alloc((size_t)BB * 64 * 64 * 4);
  float* ks1   = (float*)alloc((size_t)BB * 64 * 4);
  float* kv2   = (float*)alloc((size_t)BB * 64 * 64 * 4);
  float* ks2   = (float*)alloc((size_t)BB * 64 * 4);
  float* g1    = (float*)alloc((size_t)BB * MM1 * CH * 4);
  float* g2    = (float*)alloc((size_t)BB * MM2 * CH * 4);

  const int ROWS = BB * NPTS;                    // 64000
  k_encode<<<ROWS / 64, 128, 0, stream>>>(x, enc_w1, enc_b1, enc_w2, enc_b2, feats);
  k_fps<<<BB, 1024, 0, stream>>>(x, idx1);
  k_fuse<<<BB * MM1, 64, 0, stream>>>(x, feats, idx1, f1_w, f1_b, out1, MM1, 0.02f, 0.15f);
  k_fuse<<<BB * MM2, 64, 0, stream>>>(x, feats, idx1, f2_w, f2_b, out2, MM2, 0.05f, 0.30f);
  k_ccc<<<BB, 64, 0, stream>>>(out1, ccc_w1, ccc_b1, ccc_w2, ccc_b2, MM1);
  k_ccc<<<BB, 64, 0, stream>>>(out2, ccc_w1, ccc_b1, ccc_w2, ccc_b2, MM2);
  k_qkv<<<(BB * MM1) / 64, 128, 0, stream>>>(out1, wq, wk, wv, q1, k1, v1);
  k_qkv<<<(BB * MM2) / 64, 128, 0, stream>>>(out2, wq, wk, wv, q2, k2, v2);
  k_kvsum<<<BB, 128, 0, stream>>>(k1, v1, kv1, ks1, MM1);
  k_kvsum<<<BB, 128, 0, stream>>>(k2, v2, kv2, ks2, MM2);
  k_attnout<<<dim3(BB, MM1 / 64), 128, 0, stream>>>(out1, q1, kv1, ks1, wo, g1, MM1);
  k_attnout<<<dim3(BB, MM2 / 64), 128, 0, stream>>>(out2, q2, kv2, ks2, wo, g2, MM2);
  k_final<<<ROWS / 64, 128, 0, stream>>>(g1, g2, up_w1, up_b1, up_w2, up_b2,
                                         cls_w1, cls_b1, cls_w2, cls_b2, out);
}
